// MemoryEfficientAttention_10428180594786
// MI455X (gfx1250) — matrix operand order repaired and checked by reading, not compiled, not run on hardware
//
#include <hip/hip_runtime.h>
#include <math.h>

typedef __attribute__((ext_vector_type(16))) _Float16 v16h;
typedef __attribute__((ext_vector_type(8)))  float    v8f;
typedef __attribute__((ext_vector_type(4)))  _Float16 h4;

#define S_LEN 4096
#define D_DIM 64
#define BLK_M 64
#define BLK_N 128         // keys per iteration: amortize softmax over 2x keys
#define LDW   72          // LDS row stride for 64-wide rows (halves); 144B, 16B-aligned
#define LDK   136         // LDS row stride for 128-wide rows (halves); 272B, 16B-aligned
#define NWAVE 4
#define NBLK  (S_LEN / BLK_N)
#define LOG2E 1.4426950408889634f

// Load 16 halves (one WMMA A/B fragment slice for this lane) as two b128 LDS loads.
__device__ __forceinline__ v16h ld_frag_row(const _Float16* rowp, int off0, int off1) {
    v16h r;
    ((float4*)&r)[0] = *(const float4*)(rowp + off0);
    ((float4*)&r)[1] = *(const float4*)(rowp + off1);
    return r;
}

// Raw v_exp_f32 (2^x). Args are always <= 0; flush-to-zero on tiny results is desired.
__device__ __forceinline__ float fast_exp2(float x) {
#if __has_builtin(__builtin_amdgcn_exp2f)
    return __builtin_amdgcn_exp2f(x);
#else
    return exp2f(x);
#endif
}

// max-reduce across the 16-lane DPP row (WMMA half-group) -> all lanes get the max
__device__ __forceinline__ float rowmax16(float x) {
#if __has_builtin(__builtin_amdgcn_update_dpp)
    int t;
    t = __builtin_amdgcn_update_dpp(0, __float_as_int(x), 0x121, 0xf, 0xf, true); // row_ror:1
    x = fmaxf(x, __int_as_float(t));
    t = __builtin_amdgcn_update_dpp(0, __float_as_int(x), 0x122, 0xf, 0xf, true); // row_ror:2
    x = fmaxf(x, __int_as_float(t));
    t = __builtin_amdgcn_update_dpp(0, __float_as_int(x), 0x124, 0xf, 0xf, true); // row_ror:4
    x = fmaxf(x, __int_as_float(t));
    t = __builtin_amdgcn_update_dpp(0, __float_as_int(x), 0x128, 0xf, 0xf, true); // row_ror:8
    x = fmaxf(x, __int_as_float(t));
#else
    x = fmaxf(x, __shfl_xor(x, 1, 32));
    x = fmaxf(x, __shfl_xor(x, 2, 32));
    x = fmaxf(x, __shfl_xor(x, 4, 32));
    x = fmaxf(x, __shfl_xor(x, 8, 32));
#endif
    return x;
}

__global__ __launch_bounds__(128, 2)
void MemoryEfficientAttention_10428180594786_kernel(
        const float* __restrict__ q,
        const float* __restrict__ k,
        const float* __restrict__ v,
        const int*   __restrict__ causal_p,
        float* __restrict__ out)
{
    __shared__ _Float16 sQ [BLK_M * LDW];          // Q tile [64 q][64 d], pre-scaled (log2 dom)
    __shared__ _Float16 sK [BLK_N * LDW];          // K tile [128 key][64 d] row-major
    __shared__ _Float16 sVt[D_DIM * LDK];          // V tile transposed [64 d][128 key]
    __shared__ _Float16 sP [NWAVE * 16 * LDK];     // per-wave P scratch [16 q][128 key]

    const int tid  = threadIdx.x;
    const int wave = tid >> 5;
    const int lane = tid & 31;
    const int g    = lane >> 4;     // half-wave group
    const int ln   = lane & 15;

    const int qb = blockIdx.x;      // 64-row query block
    const int bh = blockIdx.y;      // fused batch*head
    const int causal = *causal_p;
    const float qscale = 0.125f * LOG2E;   // 1/sqrt(64) * log2(e): base-2 softmax

    const float* qg = q   + ((size_t)bh * S_LEN + (size_t)qb * BLK_M) * D_DIM;
    const float* kg = k   +  (size_t)bh * S_LEN * D_DIM;
    const float* vg = v   +  (size_t)bh * S_LEN * D_DIM;
    float*       og = out + ((size_t)bh * S_LEN + (size_t)qb * BLK_M) * D_DIM;

    // Cooperative-load geometry: float4 index fidx = i*128 + tid maps to
    // row = fidx>>4, c4 = (fidx&15)*4, and row*D_DIM + c4 == 4*fidx (linear!).
    const int t_row = tid >> 4;            // row for i=0 slot (rows advance by 8 per i)
    const int t_c4  = (tid & 15) << 2;

    // ---- cooperative Q tile load: fp32 -> f16 (scale folded in), linear addressing ----
    {
        const float* qt = qg + (size_t)tid * 4;    // + i*512 floats folded to ioffset
#pragma unroll
        for (int i = 0; i < 8; ++i) {
            float4 d = *(const float4*)(qt + (size_t)i * 512);
            h4 hv = { (_Float16)(d.x * qscale), (_Float16)(d.y * qscale),
                      (_Float16)(d.z * qscale), (_Float16)(d.w * qscale) };
            *(h4*)(&sQ[(t_row + i * 8) * LDW + t_c4]) = hv;
        }
    }
    __syncthreads();

    // Q A-fragments for this wave's 16 rows (loop invariant)
    const _Float16* qrow = &sQ[(wave * 16 + ln) * LDW];
    const v16h aQ0 = ld_frag_row(qrow, g * 8,      16 + g * 8);   // d = 0..31
    const v16h aQ1 = ld_frag_row(qrow, 32 + g * 8, 48 + g * 8);   // d = 32..63

    // all-ones B fragment: P @ ones => row-sums in every column of the D tile
    v16h ones;
#pragma unroll
    for (int e = 0; e < 16; ++e) ones[e] = (_Float16)1.0f;

    v8f Ofr[4] = {v8f{}, v8f{}, v8f{}, v8f{}};
    v8f Lacc = {};
    float m_st[8];
#pragma unroll
    for (int r = 0; r < 8; ++r) m_st[r] = -INFINITY;

    const int jmax = causal ? (qb >> 1) : (NBLK - 1);
    _Float16* sPw = &sP[wave * 16 * LDK];

    for (int j = 0; j <= jmax; ++j) {
        __syncthreads();   // all waves done reading previous sK/sVt
        // ---- load K (row-major) + V (transposed) 128-key tile, fp32 -> f16 ----
        {
            const float* kt = kg + ((size_t)j << 13) + (size_t)tid * 4;  // j*8192 floats
            const float* vt = vg + ((size_t)j << 13) + (size_t)tid * 4;
#pragma unroll
            for (int i = 0; i < 16; ++i) {
                int row = t_row + i * 8;
                float4 dk = *(const float4*)(kt + (size_t)i * 512);
                h4 hk = { (_Float16)dk.x, (_Float16)dk.y, (_Float16)dk.z, (_Float16)dk.w };
                *(h4*)(&sK[row * LDW + t_c4]) = hk;
                float4 dv = *(const float4*)(vt + (size_t)i * 512);
                sVt[(t_c4 + 0) * LDK + row] = (_Float16)dv.x;
                sVt[(t_c4 + 1) * LDK + row] = (_Float16)dv.y;
                sVt[(t_c4 + 2) * LDK + row] = (_Float16)dv.z;
                sVt[(t_c4 + 3) * LDK + row] = (_Float16)dv.w;
            }
        }
        __syncthreads();

        // ---- prefetch next tile into cache: one base each, line offsets in ioffset ----
        if (j < jmax) {
            const char* kp = (const char*)kg + ((size_t)(j + 1) << 15) + (size_t)tid * 64;
            const char* vp = (const char*)vg + ((size_t)(j + 1) << 15) + (size_t)tid * 64;
            __builtin_prefetch(kp,         0, 3);
            __builtin_prefetch(kp +  8192, 0, 3);
            __builtin_prefetch(kp + 16384, 0, 3);
            __builtin_prefetch(kp + 24576, 0, 3);
            __builtin_prefetch(vp,         0, 3);
            __builtin_prefetch(vp +  8192, 0, 3);
            __builtin_prefetch(vp + 16384, 0, 3);
            __builtin_prefetch(vp + 24576, 0, 3);
        }

        // ---- S = (Q*scale*log2e) @ K^T : 8 N-tiles, d-depth 64 = 2 chained WMMAs ----
        v8f sf[8];
#pragma unroll
        for (int nt = 0; nt < 8; ++nt) {
            const _Float16* krow = &sK[(nt * 16 + ln) * LDW];
            v16h bk0 = ld_frag_row(krow, 8 * g,      16 + 8 * g);       // d = 0..31
            v16h bk1 = ld_frag_row(krow, 32 + 8 * g, 48 + 8 * g);  // d = 32..63
            v8f acc = {};
            acc = __builtin_amdgcn_wmma_f32_16x16x32_f16(false, aQ0, false, bk0,
                                                         (short)0, acc, false, false);
            acc = __builtin_amdgcn_wmma_f32_16x16x32_f16(false, aQ1, false, bk1,
                                                         (short)0, acc, false, false);
            sf[nt] = acc;
        }

        // ---- causal mask on diagonal block ----
        if (causal && (j == jmax)) {
#pragma unroll
            for (int nt = 0; nt < 8; ++nt) {
                int kglob = j * BLK_N + nt * 16 + ln;
#pragma unroll
                for (int r = 0; r < 8; ++r) {
                    int qglob = qb * BLK_M + wave * 16 + g * 8 + r;
                    sf[nt][r] = (kglob <= qglob) ? sf[nt][r] : -INFINITY;
                }
            }
        }

        // ---- online softmax (base-2): rowmax via DPP, P = exp2(s - m) ----
        float alpha[8];
#pragma unroll
        for (int r = 0; r < 8; ++r) {
            float mx = fmaxf(fmaxf(fmaxf(sf[0][r], sf[1][r]), fmaxf(sf[2][r], sf[3][r])),
                             fmaxf(fmaxf(sf[4][r], sf[5][r]), fmaxf(sf[6][r], sf[7][r])));
            mx = rowmax16(mx);
            float m_new = fmaxf(m_st[r], mx);
            alpha[r] = fast_exp2(m_st[r] - m_new);
            m_st[r] = m_new;
#pragma unroll
            for (int nt = 0; nt < 8; ++nt)
                sf[nt][r] = fast_exp2(sf[nt][r] - m_new);
        }

        // ---- P: D-layout -> A-layout via per-wave LDS tile ----
#pragma unroll
        for (int nt = 0; nt < 8; ++nt)
#pragma unroll
            for (int r = 0; r < 8; ++r)
                sPw[(g * 8 + r) * LDK + nt * 16 + ln] = (_Float16)sf[nt][r];
        asm volatile("s_wait_dscnt 0x0" ::: "memory");   // wave-local store->load ordering

        const _Float16* prow = &sPw[ln * LDK];
        v16h aP[4];
#pragma unroll
        for (int kc = 0; kc < 4; ++kc)
            aP[kc] = ld_frag_row(prow, kc * 32 + g * 8, kc * 32 + 16 + g * 8);

        // ---- L = L*alpha + P @ ones  (row-sum on the matrix pipe) ----
#pragma unroll
        for (int r = 0; r < 8; ++r) Lacc[r] *= alpha[r];
#pragma unroll
        for (int kc = 0; kc < 4; ++kc)
            Lacc = __builtin_amdgcn_wmma_f32_16x16x32_f16(false, aP[kc], false, ones,
                                                          (short)0, Lacc, false, false);

        // ---- O = O*alpha + P @ V ----
#pragma unroll
        for (int dt = 0; dt < 4; ++dt) {
#pragma unroll
            for (int r = 0; r < 8; ++r) Ofr[dt][r] *= alpha[r];
            const _Float16* vrow = &sVt[(dt * 16 + ln) * LDK];
#pragma unroll
            for (int kc = 0; kc < 4; ++kc) {
                v16h bv = ld_frag_row(vrow, kc * 32 + 8 * g, kc * 32 + 16 + 8 * g);
                Ofr[dt] = __builtin_amdgcn_wmma_f32_16x16x32_f16(false, aP[kc], false, bv,
                                                                 (short)0, Ofr[dt], false, false);
            }
        }
    }

    // ---- epilogue: normalize and store fp32 ----
#pragma unroll
    for (int r = 0; r < 8; ++r) {
        float invl = 1.0f / Lacc[r];
        int row = wave * 16 + g * 8 + r;
#pragma unroll
        for (int dt = 0; dt < 4; ++dt)
            og[(size_t)row * D_DIM + dt * 16 + ln] = Ofr[dt][r] * invl;
    }
}

extern "C" void kernel_launch(void* const* d_in, const int* in_sizes, int n_in,
                              void* d_out, int out_size, void* d_ws, size_t ws_size,
                              hipStream_t stream) {
    const float* q = (const float*)d_in[0];
    const float* k = (const float*)d_in[1];
    const float* v = (const float*)d_in[2];
    const int* causal = (const int*)d_in[3];
    float* out = (float*)d_out;
    int bh = in_sizes[0] / (S_LEN * D_DIM);   // B*H = 64
    dim3 grid(S_LEN / BLK_M, bh);
    MemoryEfficientAttention_10428180594786_kernel<<<grid, 128, 0, stream>>>(q, k, v, causal, out);
}
